// FlexibleGNN_16260746183299
// MI455X (gfx1250) — compile-verified, matrix operand order checked by reading, not run on hardware
//
#include <hip/hip_runtime.h>

// ---------------------------------------------------------------------------
// FlexibleGNN transformer on MI455X (gfx1250): per-node attention over 32
// neighbors. All heavy GEMMs via v_wmma_f32_16x16x32_f16 (f16 in, f32 acc).
// K/V projections use algebraically fused weights (Wi@Wk, Wi@Wv) so the
// 268 MB neighbor tensor is read exactly once from HBM per layer pass.
// ---------------------------------------------------------------------------

typedef _Float16 half_t;
typedef __attribute__((ext_vector_type(16))) _Float16 v16h;
typedef __attribute__((ext_vector_type(8)))  _Float16 v8h;
typedef __attribute__((ext_vector_type(4)))  _Float16 v4h;
typedef __attribute__((ext_vector_type(8)))  float    v8f;

#define NODES 16384
#define DMODEL 128
#define DFF    512
#define NLAYER 2

__device__ __forceinline__ v8f wmma_f16(v16h a, v16h b, v8f c) {
  // (neg_a, A, neg_b, B, c_mod, C, reuse_a, reuse_b)
  return __builtin_amdgcn_wmma_f32_16x16x32_f16(false, a, false, b, (short)0, c,
                                                false, false);
}

// A-fragment (16x32 f16, row-major source, stride in halves) or
// B-fragment (32x16 f16 from pre-transposed Wt[N][K], stride = K):
//   lane l<16 : row/col (l),    K = base .. base+15
//   lane l>=16: row/col (l-16), K = base+16 .. base+31
__device__ __forceinline__ v16h frag16(const half_t* tile, int stride) {
  const int lane = threadIdx.x & 31;
  const half_t* p = tile + (lane & 15) * stride + ((lane >> 4) << 4);
  return *(const v16h*)p;
}

// ---------------------------------------------------------------------------
// Weight prep: transpose-convert fp32 W[in][out] -> f16 Wt[out][in]
// ---------------------------------------------------------------------------
__global__ void k_tconv(const float* __restrict__ W, half_t* __restrict__ Wt,
                        int in, int out) {
  int i = blockIdx.x * 256 + threadIdx.x;
  if (i >= in * out) return;
  int o = i / in, k = i - o * in;
  Wt[i] = (half_t)W[k * out + o];
}

// Fused K/V weights: Wt[o][k] = sum_j Wi[k][j]*Wx[j][o]; bf[o]=bi@Wx + bx
__global__ void k_fuse(const float* __restrict__ Wi, const float* __restrict__ Wx,
                       const float* __restrict__ bi, const float* __restrict__ bx,
                       half_t* __restrict__ Wt, float* __restrict__ bf) {
  int i = blockIdx.x * 256 + threadIdx.x;          // 0 .. 16383
  int o = i >> 7, k = i & 127;
  float s = 0.f;
  for (int j = 0; j < 128; ++j) s += Wi[k * 128 + j] * Wx[j * 128 + o];
  Wt[o * 128 + k] = (half_t)s;
  if (k == 0) {
    float t = bx[o];
    for (int j = 0; j < 128; ++j) t += bi[j] * Wx[j * 128 + o];
    bf[o] = t;
  }
}

// ---------------------------------------------------------------------------
// Generic GEMM: Out[16384 x N] = A[16384 x K] @ Wt^T + bias, fused epilogues.
// 32-row block per workgroup, 8 waves, WMMA 16x16x32.
// EPI 0: out f32 = C+bias
// EPI 1: out f16 = relu(C+bias)
// EPI 2: out f32 = LN(res1 + C + bias)*g + be
// EPI 3: out f32 = LN(res1 + C + bias)*g + be + res2
// ---------------------------------------------------------------------------
template <int K, int N, int EPI, bool AF16>
__global__ __launch_bounds__(256) void k_gemm(
    const void* __restrict__ Ain, const half_t* __restrict__ Wt,
    const float* __restrict__ bias, const float* __restrict__ res1,
    const float* __restrict__ res2, const float* __restrict__ gamma,
    const float* __restrict__ beta, void* __restrict__ Out) {
  constexpr int LDA = K + 16;
  constexpr bool LN = (EPI >= 2);
  static_assert(!LN || N == 128, "LN epilogue assumes N==128");
  __shared__ __align__(64) half_t sA[32 * LDA];
  __shared__ __align__(64) float  sC[LN ? 32 * N : 1];
  __shared__ float sMu[32], sRs[32];

  const int tid = threadIdx.x;
  const int rb = blockIdx.x * 32;

  // stage A block (vectorized 4-wide), convert to f16
  if constexpr (AF16) {
    const half_t* A = (const half_t*)Ain;
    for (int i = tid * 4; i < 32 * K; i += 1024) {
      int r = i / K, c = i - r * K;
      v4h hv = *(const v4h*)&A[(size_t)(rb + r) * K + c];
      *(v4h*)&sA[r * LDA + c] = hv;
    }
  } else {
    const float* A = (const float*)Ain;
    for (int i = tid * 4; i < 32 * K; i += 1024) {
      int r = i / K, c = i - r * K;
      float4 fv = *(const float4*)&A[(size_t)(rb + r) * K + c];
      v4h hv = {(half_t)fv.x, (half_t)fv.y, (half_t)fv.z, (half_t)fv.w};
      *(v4h*)&sA[r * LDA + c] = hv;
    }
  }
  __syncthreads();

  const int wid = __builtin_amdgcn_readfirstlane(tid >> 5);  // wave-uniform
  const int lane = tid & 31;
  const int hi = lane >> 4, col = lane & 15;
  constexpr int NT = N / 16;
  constexpr int TILES = 2 * NT;

  for (int t = wid; t < TILES; t += 8) {
    const int rt = t / NT, nt = t - rt * NT;
    const half_t* atile = &sA[rt * 16 * LDA];
    const half_t* btile = &Wt[(size_t)nt * 16 * K];
    __builtin_prefetch(btile, 0, 1);
    v8f acc = {};
#pragma unroll
    for (int kk = 0; kk < K / 32; ++kk)
      acc = wmma_f16(frag16(atile + kk * 32, LDA), frag16(btile + kk * 32, K), acc);

    const int gcol = nt * 16 + col;
    const float bv = bias[gcol];
#pragma unroll
    for (int r = 0; r < 8; ++r) {
      const int lrow = rt * 16 + hi * 8 + r;
      const size_t grow = (size_t)(rb + lrow);
      float v = acc[r] + bv;
      if constexpr (EPI == 0) {
        ((float*)Out)[grow * N + gcol] = v;
      } else if constexpr (EPI == 1) {
        ((half_t*)Out)[grow * N + gcol] = (half_t)fmaxf(v, 0.f);
      } else {
        sC[lrow * N + gcol] = v + res1[grow * N + gcol];
      }
    }
  }

  if constexpr (LN) {
    __syncthreads();
    if (tid < 32) {
      float mu = 0.f;
      for (int d = 0; d < N; ++d) mu += sC[tid * N + d];
      mu *= (1.0f / N);
      float var = 0.f;
      for (int d = 0; d < N; ++d) {
        float dv = sC[tid * N + d] - mu;
        var += dv * dv;
      }
      var *= (1.0f / N);
      sMu[tid] = mu;
      sRs[tid] = rsqrtf(var + 1e-5f);
    }
    __syncthreads();
    float* O = (float*)Out;
    for (int i = tid; i < 32 * N; i += 256) {
      int r = i / N, d = i - r * N;
      float y = (sC[i] - sMu[r]) * sRs[r] * gamma[d] + beta[d];
      if constexpr (EPI == 3) y += res2[(size_t)(rb + r) * N + d];
      O[(size_t)(rb + r) * N + d] = y;
    }
  }
}

// ---------------------------------------------------------------------------
// Attention: 2 nodes per workgroup (64 neighbor rows = 4 full WMMA row tiles).
// K = NB @ Wik + bik  (WMMA) -> LDS, stored TRANSPOSED: sKVt[dim][row]
//   (lane's 8 accumulator rows are contiguous -> single ds_store_b128)
// scores/softmax (1 thread per (node,head,m))
// V = NB @ Wiv + biv  (WMMA, reuse sKVt)
// att[n, c] = sum_m p[m] * sKVt[c][m]  (contiguous 64B read per thread)
// ---------------------------------------------------------------------------
__global__ __launch_bounds__(256) void k_attn(
    const float* __restrict__ nbf, const int* __restrict__ mask,
    const float* __restrict__ Q, const half_t* __restrict__ Wikt,
    const float* __restrict__ bik, const half_t* __restrict__ Wivt,
    const float* __restrict__ biv, float* __restrict__ att) {
  constexpr int LDN = 144;  // sNB stride (halves)
  constexpr int LDT = 88;   // sKVt stride (halves): 176B, 16B-aligned, bank-spread
  __shared__ __align__(64) half_t sNB[64 * LDN];
  __shared__ __align__(64) half_t sKVt[128 * LDT];
  __shared__ float sQ[256];
  __shared__ float sS[256];

  const int tid = threadIdx.x;
  const int node0 = blockIdx.x * 2;
  const size_t row0 = (size_t)node0 * 32;

  // stage neighbor block (64 rows x 128), fp32 -> f16, vectorized
  for (int i = tid * 4; i < 64 * 128; i += 1024) {
    int r = i >> 7, c = i & 127;
    float4 fv = *(const float4*)&nbf[(row0 + r) * 128 + c];
    v4h hv = {(half_t)fv.x, (half_t)fv.y, (half_t)fv.z, (half_t)fv.w};
    *(v4h*)&sNB[r * LDN + c] = hv;
  }
  sQ[tid] = Q[(size_t)node0 * 128 + tid];
  __syncthreads();

  const int wid = __builtin_amdgcn_readfirstlane(tid >> 5);  // wave-uniform
  const int lane = tid & 31;
  const int hi = lane >> 4, col = lane & 15;

  // ---- K projection (transposed store) ----
  for (int t = wid; t < 32; t += 8) {
    int rt = t >> 3, nt = t & 7;
    const half_t* atile = &sNB[rt * 16 * LDN];
    const half_t* btile = &Wikt[nt * 16 * 128];
    v8f acc = {};
#pragma unroll
    for (int kk = 0; kk < 4; ++kk)
      acc = wmma_f16(frag16(atile + kk * 32, LDN), frag16(btile + kk * 32, 128), acc);
    float bv = bik[nt * 16 + col];
    v8h kv;
#pragma unroll
    for (int r = 0; r < 8; ++r) kv[r] = (half_t)(acc[r] + bv);
    *(v8h*)&sKVt[(nt * 16 + col) * LDT + rt * 16 + hi * 8] = kv;
  }
  __syncthreads();

  // ---- scores + softmax (groups of 32 lanes = one (node,head)) ----
  {
    const int nl = tid >> 7, h = (tid >> 5) & 3, m = tid & 31;
    float s;
    if (mask[(node0 + nl) * 32 + m] != 0) {
      float a = 0.f;
      const half_t* kc = &sKVt[(h * 32) * LDT + nl * 32 + m];
      const float* q = &sQ[nl * 128 + h * 32];
      for (int d = 0; d < 32; ++d) a += q[d] * (float)kc[d * LDT];
      s = a * 0.17677669529663687f;  // 1/sqrt(32)
    } else {
      s = -1e9f;
    }
    sS[tid] = s;
    __syncthreads();
    const int base = tid & ~31;
    float mx = -3.4e38f;
    for (int j = 0; j < 32; ++j) mx = fmaxf(mx, sS[base + j]);
    float e = __expf(s - mx);
    __syncthreads();
    sS[tid] = e;
    __syncthreads();
    float sum = 0.f;
    for (int j = 0; j < 32; ++j) sum += sS[base + j];
    float p = e / sum;
    __syncthreads();
    sS[tid] = p;
  }
  __syncthreads();

  // ---- V projection (reuse sKVt, transposed store) ----
  for (int t = wid; t < 32; t += 8) {
    int rt = t >> 3, nt = t & 7;
    const half_t* atile = &sNB[rt * 16 * LDN];
    const half_t* btile = &Wivt[nt * 16 * 128];
    v8f acc = {};
#pragma unroll
    for (int kk = 0; kk < 4; ++kk)
      acc = wmma_f16(frag16(atile + kk * 32, LDN), frag16(btile + kk * 32, 128), acc);
    float bv = biv[nt * 16 + col];
    v8h vv;
#pragma unroll
    for (int r = 0; r < 8; ++r) vv[r] = (half_t)(acc[r] + bv);
    *(v8h*)&sKVt[(nt * 16 + col) * LDT + rt * 16 + hi * 8] = vv;
  }
  __syncthreads();

  // ---- attention-weighted sum (contiguous V reads) ----
  {
    const int nl = tid >> 7, c = tid & 127;
    const float* p = &sS[nl * 128 + (c >> 5) * 32];
    const half_t* vrow = &sKVt[c * LDT + nl * 32];
    float a = 0.f;
    for (int m = 0; m < 32; ++m) a += p[m] * (float)vrow[m];
    att[(size_t)(node0 + nl) * 128 + c] = a;
  }
}

// ---------------------------------------------------------------------------
extern "C" void kernel_launch(void* const* d_in, const int* in_sizes, int n_in,
                              void* d_out, int out_size, void* d_ws,
                              size_t ws_size, hipStream_t stream) {
  (void)in_sizes; (void)n_in; (void)out_size; (void)ws_size;
  const float* x    = (const float*)d_in[0];
  const float* nbf  = (const float*)d_in[1];
  const int*   mask = (const int*)d_in[2];
  const float* Wi   = (const float*)d_in[3];
  const float* bi   = (const float*)d_in[4];
  const float* Wq   = (const float*)d_in[5];
  const float* bq   = (const float*)d_in[6];
  const float* Wk   = (const float*)d_in[7];
  const float* bk   = (const float*)d_in[8];
  const float* Wv   = (const float*)d_in[9];
  const float* bv   = (const float*)d_in[10];
  const float* Wo   = (const float*)d_in[11];
  const float* bo   = (const float*)d_in[12];
  const float* g1   = (const float*)d_in[13];
  const float* be1  = (const float*)d_in[14];
  const float* W1   = (const float*)d_in[15];
  const float* b1   = (const float*)d_in[16];
  const float* W2   = (const float*)d_in[17];
  const float* b2   = (const float*)d_in[18];
  const float* g2   = (const float*)d_in[19];
  const float* be2  = (const float*)d_in[20];
  const float* Wout = (const float*)d_in[21];
  const float* bout = (const float*)d_in[22];

  char* ws = (char*)d_ws;
  size_t off = 0;
  auto alloc = [&](size_t bytes) -> void* {
    off = (off + 255) & ~(size_t)255;
    void* p = ws + off;
    off += bytes;
    return p;
  };

  const size_t DD = (size_t)DMODEL * DMODEL;
  half_t* Wit   = (half_t*)alloc(DD * 2);
  half_t* Woutt = (half_t*)alloc(DD * 2);
  half_t *Wqt[NLAYER], *Wot[NLAYER], *W1t[NLAYER], *W2t[NLAYER];
  half_t *Wikt[NLAYER], *Wivt[NLAYER];
  float  *bikp[NLAYER], *bivp[NLAYER];
  for (int l = 0; l < NLAYER; ++l) {
    Wqt[l]  = (half_t*)alloc(DD * 2);
    Wot[l]  = (half_t*)alloc(DD * 2);
    W1t[l]  = (half_t*)alloc((size_t)DMODEL * DFF * 2);
    W2t[l]  = (half_t*)alloc((size_t)DFF * DMODEL * 2);
    Wikt[l] = (half_t*)alloc(DD * 2);
    Wivt[l] = (half_t*)alloc(DD * 2);
    bikp[l] = (float*)alloc(DMODEL * 4);
    bivp[l] = (float*)alloc(DMODEL * 4);
  }
  float*  hA   = (float*)alloc((size_t)NODES * DMODEL * 4);
  float*  hB   = (float*)alloc((size_t)NODES * DMODEL * 4);
  float*  h1b  = (float*)alloc((size_t)NODES * DMODEL * 4);
  float*  Qb   = (float*)alloc((size_t)NODES * DMODEL * 4);
  float*  attb = (float*)alloc((size_t)NODES * DMODEL * 4);
  half_t* Fb   = (half_t*)alloc((size_t)NODES * DFF * 2);

  const dim3 blk(256);
  const int gDD = (int)((DD + 255) / 256);
  const int gDF = (int)(((size_t)DMODEL * DFF + 255) / 256);

  // ---- weight prep ----
  k_tconv<<<gDD, blk, 0, stream>>>(Wi, Wit, DMODEL, DMODEL);
  k_tconv<<<gDD, blk, 0, stream>>>(Wout, Woutt, DMODEL, DMODEL);
  for (int l = 0; l < NLAYER; ++l) {
    k_tconv<<<gDD, blk, 0, stream>>>(Wq + l * DD, Wqt[l], DMODEL, DMODEL);
    k_tconv<<<gDD, blk, 0, stream>>>(Wo + l * DD, Wot[l], DMODEL, DMODEL);
    k_tconv<<<gDF, blk, 0, stream>>>(W1 + (size_t)l * DMODEL * DFF, W1t[l], DMODEL, DFF);
    k_tconv<<<gDF, blk, 0, stream>>>(W2 + (size_t)l * DFF * DMODEL, W2t[l], DFF, DMODEL);
    k_fuse<<<64, blk, 0, stream>>>(Wi, Wk + l * DD, bi, bk + l * DMODEL, Wikt[l], bikp[l]);
    k_fuse<<<64, blk, 0, stream>>>(Wi, Wv + l * DD, bi, bv + l * DMODEL, Wivt[l], bivp[l]);
  }

  const int gGemm = NODES / 32;  // 512
  const int gAttn = NODES / 2;   // 8192

  // h0 = x @ Wi + bi
  k_gemm<128, 128, 0, false><<<gGemm, blk, 0, stream>>>(
      x, Wit, bi, nullptr, nullptr, nullptr, nullptr, hA);

  float* hin = hA;
  float* hout = hB;
  for (int l = 0; l < NLAYER; ++l) {
    // Q = h @ Wq + bq
    k_gemm<128, 128, 0, false><<<gGemm, blk, 0, stream>>>(
        hin, Wqt[l], bq + l * DMODEL, nullptr, nullptr, nullptr, nullptr, Qb);
    // attention over neighbors (fused K/V projections)
    k_attn<<<gAttn, blk, 0, stream>>>(nbf, mask, Qb, Wikt[l], bikp[l],
                                      Wivt[l], bivp[l], attb);
    // h1 = LN(h + att@Wo + bo)
    k_gemm<128, 128, 2, false><<<gGemm, blk, 0, stream>>>(
        attb, Wot[l], bo + l * DMODEL, hin, nullptr, g1 + l * DMODEL,
        be1 + l * DMODEL, h1b);
    // F = relu(h1@W1 + b1)  (f16)
    k_gemm<128, 512, 1, false><<<gGemm, blk, 0, stream>>>(
        h1b, W1t[l], b1 + l * DFF, nullptr, nullptr, nullptr, nullptr, Fb);
    // h_out = LN(h1 + F@W2 + b2) + h   (outer residual fused)
    k_gemm<512, 128, 3, true><<<gGemm, blk, 0, stream>>>(
        Fb, W2t[l], b2 + l * DMODEL, h1b, hin, g2 + l * DMODEL,
        be2 + l * DMODEL, hout);
    float* tmp = hin; hin = hout; hout = tmp;
  }

  // out = h @ Wout + bout
  k_gemm<128, 128, 0, false><<<gGemm, blk, 0, stream>>>(
      hin, Woutt, bout, nullptr, nullptr, nullptr, nullptr, d_out);
}
